// InputStream_15109694947667
// MI455X (gfx1250) — matrix operand order repaired and checked by reading, not compiled, not run on hardware
//
#include <hip/hip_runtime.h>

typedef __attribute__((ext_vector_type(16))) __bf16        v16bf;
typedef __attribute__((ext_vector_type(8)))  float         v8f;
typedef __attribute__((ext_vector_type(4)))  float         f32x4;
typedef __attribute__((ext_vector_type(2)))  float         v2f;
typedef __attribute__((ext_vector_type(4)))  unsigned int  u32x4;
typedef __attribute__((ext_vector_type(2)))  unsigned int  u32x2;

union ABu { v16bf v; u32x4 q[2]; };

__device__ __forceinline__ unsigned short f2bf(float f) {
    unsigned int u = __float_as_uint(f);
    u += 0x7FFFu + ((u >> 16) & 1u);           // round-to-nearest-even
    return (unsigned short)(u >> 16);
}
__device__ __forceinline__ unsigned int pack2bf(float lo, float hi) {
    return (unsigned int)f2bf(lo) | ((unsigned int)f2bf(hi) << 16);
}
__device__ __forceinline__ float sigm(float x)  { return 1.f / (1.f + __expf(-x)); }
__device__ __forceinline__ float lrelu(float x) { return x > 0.f ? x : 0.3f * x; }

// ---------------------------------------------------------------------------
// Pack conv weights (wx: 3x3xCINx16, wh: 3x3x4x16) into exact WMMA B-fragment
// layout for v_wmma_f32_16x16x32_bf16.  K order: tap*8 + ch, ch = [x(0..CIN-1),
// h(CIN..CIN+3), pad], K padded to 96 (3 k-steps of 32).
// B layout (32x16, 16-bit): lane half h, elem j<8 -> K=8h+j,
// elem j>=8 -> K=16+8h+(j-8).
// Output: bfrag[s][lane][j]  (3*32*16 bf16)
// ---------------------------------------------------------------------------
__global__ void pack_bfrag(const float* __restrict__ wx, const float* __restrict__ wh,
                           int cin, unsigned short* __restrict__ out) {
    int idx = blockIdx.x * blockDim.x + threadIdx.x;
    if (idx >= 3 * 32 * 16) return;
    int j    = idx & 15;
    int lane = (idx >> 4) & 31;
    int s    = idx >> 9;
    int K    = s * 32 + (j >> 3) * 16 + (lane >> 4) * 8 + (j & 7);
    int n    = lane & 15;
    float val = 0.f;
    if (K < 72) {
        int tap = K >> 3, ch = K & 7;
        if (ch < cin)           val = wx[(tap * cin + ch) * 16 + n];
        else if (ch < cin + 4)  val = wh[(tap * 4 + (ch - cin)) * 16 + n];
    }
    out[idx] = f2bf(val);
}

// ---------------------------------------------------------------------------
// Fused ConvLSTM cell step.  Grid (W/16, H/8, B), block 256 (8 waves).
// Wave w handles row by*8+w, cols bx*16..+15: one 16x16 z-tile via 3 bf16 WMMA.
// Then gates + state update + BN-output write.
// ---------------------------------------------------------------------------
template<int CIN, bool BN_F32>
__global__ __launch_bounds__(256) void convlstm_cell(
    const void* __restrict__ x_t, long xBatch,           // per-batch element stride
    const float* __restrict__ h_in, float* __restrict__ h_out,
    float* __restrict__ c_st,
    void* __restrict__ bn_out, long bnBatch,
    const unsigned short* __restrict__ bfrag,
    const float* __restrict__ bias16,
    const float* __restrict__ gam, const float* __restrict__ bet,
    const float* __restrict__ mu,  const float* __restrict__ var,
    int writeBn)
{
    __shared__ __align__(16) unsigned short stage[10 * 18 * 8];   // halo tile, bf16
    __shared__ float zb[8][16][16];                               // per-wave z exchange

    const int bx = blockIdx.x;
    const int by = blockIdx.y;
    const int b  = blockIdx.z;
    const int tid  = threadIdx.x;
    const int wave = tid >> 5;
    const int lane = tid & 31;

    // ---- stage 10x18 halo (x-channels + h-channels, zero-padded to 8) ----
    // one b128 global load for h, one b128 LDS store per pixel
    for (int i = tid; i < 10 * 18; i += 256) {
        int lr = i / 18, lc = i % 18;
        int gr = by * 8 - 1 + lr;
        int gc = bx * 16 - 1 + lc;
        bool in = (gr >= 0) & (gr < 128) & (gc >= 0) & (gc < 128);
        u32x4 q = {0u, 0u, 0u, 0u};
        if (in) {
            long pofs = (long)gr * 128 + gc;
            f32x4 h4 = *(const f32x4*)(h_in + (((long)b * 128 + gr) * 128 + gc) * 4);
            if constexpr (CIN == 3) {
                const float* xp = (const float*)x_t + (long)b * xBatch + pofs * 3;
                float x0 = xp[0], x1 = xp[1], x2 = xp[2];
                q.x = pack2bf(x0, x1);
                q.y = pack2bf(x2, h4.x);
                q.z = pack2bf(h4.y, h4.z);
                q.w = (unsigned int)f2bf(h4.w);          // slot7 = 0 pad
            } else {
                const unsigned short* xp = (const unsigned short*)x_t + (long)b * xBatch + pofs * 4;
                u32x2 xq = *(const u32x2*)xp;            // 4 bf16 x-channels
                q.x = xq.x;
                q.y = xq.y;
                q.z = pack2bf(h4.x, h4.y);
                q.w = pack2bf(h4.z, h4.w);
            }
        }
        *(u32x4*)&stage[i * 8] = q;
    }
    __syncthreads();

    // ---- z(16px x 16gatech) = A(16x96) x B(96x16) via 3 bf16 WMMA ----
    const int m     = lane & 15;
    const int khalf = lane >> 4;
    v8f acc = {};
    const u32x4 zq = {0u, 0u, 0u, 0u};
#pragma unroll
    for (int s = 0; s < 3; ++s) {
        ABu B;
        const unsigned short* bp = bfrag + ((s * 32 + lane) << 4);
        B.q[0] = *(const u32x4*)(bp);
        B.q[1] = *(const u32x4*)(bp + 8);

        int tap0 = s * 4 + khalf;        // A elems 0..7  : K = s*32 + khalf*8 + j
        int tap1 = tap0 + 2;             // A elems 8..15 : K = s*32 + 16 + khalf*8 + j
        ABu A;
        if (tap0 < 9) {
            int lr = wave + tap0 / 3, lc = m + tap0 % 3;
            A.q[0] = *(const u32x4*)&stage[(lr * 18 + lc) * 8];
        } else A.q[0] = zq;
        if (tap1 < 9) {
            int lr = wave + tap1 / 3, lc = m + tap1 % 3;
            A.q[1] = *(const u32x4*)&stage[(lr * 18 + lc) * 8];
        } else A.q[1] = zq;

        acc = __builtin_amdgcn_wmma_f32_16x16x32_bf16(false, A.v, false, B.v,
                                                      (short)0, acc, false, false);
    }

    // ---- z + bias into LDS for cross-lane gate regrouping ----
    float bn_ = bias16[lane & 15];
#pragma unroll
    for (int v = 0; v < 8; ++v)
        zb[wave][v + 8 * khalf][lane & 15] = acc[v] + bn_;
    __syncthreads();

    // ---- gates i,f,g,o ; state update ; BN output ----
#pragma unroll
    for (int e = 0; e < 2; ++e) {
        int s   = lane + e * 32;            // 64 states per wave (16 px x 4 fch)
        int mm2 = s >> 2;
        int fc  = s & 3;
        float zi = zb[wave][mm2][fc];
        float zf = zb[wave][mm2][fc + 4];
        float zg = zb[wave][mm2][fc + 8];
        float zo = zb[wave][mm2][fc + 12];
        int gr = by * 8 + wave;
        int gc = bx * 16 + mm2;
        long idx = (((long)b * 128 + gr) * 128 + gc) * 4 + fc;
        float co = c_st[idx];
        float ig = sigm(zi);
        float fg = sigm(zf);
        float gg = tanhf(zg);
        float og = sigm(zo);
        float cn = fg * co + ig * gg;
        float hn = og * tanhf(cn);
        c_st[idx]  = cn;
        h_out[idx] = hn;
        if (writeBn) {
            float rs = rsqrtf(var[fc] + 1e-3f);
            float aa = gam[fc] * rs;
            float dd = bet[fc] - mu[fc] * aa;
            float bv = aa * hn + dd;
            long oidx = (long)b * bnBatch + ((long)gr * 128 + gc) * 4 + fc;
            if constexpr (BN_F32) ((float*)bn_out)[oidx] = bv;
            else ((unsigned short*)bn_out)[oidx] = f2bf(bv);
        }
    }
}

// ---------------------------------------------------------------------------
// Dense1: (64 x 65536) @ (65536 x 128), K split over 512 blocks of 128.
// fp32 WMMA 16x16x4.  LeakyReLU applied to flatten input while staging.
// Writes partial[blk][64][128].
// ---------------------------------------------------------------------------
__global__ __launch_bounds__(256) void dense1(const float* __restrict__ act,
                                              const float* __restrict__ w,
                                              float* __restrict__ partial)
{
    __shared__ __align__(16) float sact[64][132];   // 528B pitch: 16B-aligned rows
    const int blk  = blockIdx.x;            // K chunk of 128
    const int tid  = threadIdx.x;
    const int wave = tid >> 5, lane = tid & 31;
    const long k0  = (long)blk * 128;

    for (int i = tid; i < 64 * 32; i += 256) {      // 4 floats per thread-iter
        int mrow = i >> 5, kk4 = (i & 31) * 4;
        f32x4 a = *(const f32x4*)(act + (long)mrow * 65536 + k0 + kk4);
        a.x = lrelu(a.x); a.y = lrelu(a.y); a.z = lrelu(a.z); a.w = lrelu(a.w);
        *(f32x4*)&sact[mrow][kk4] = a;
    }
    __syncthreads();

    const int nloc = lane & 15;
    const int kh   = lane >> 4;
    const int n    = wave * 16 + nloc;
    v8f a0c = {}, a1c = {}, a2c = {}, a3c = {};
    for (int kk = 0; kk < 128; kk += 4) {
        int kl = kk + 2 * kh;
        long k = k0 + kl;
        v2f bf = { w[k * 128 + n], w[(k + 1) * 128 + n] };
        v2f a0 = { sact[ 0 + nloc][kl], sact[ 0 + nloc][kl + 1] };
        v2f a1 = { sact[16 + nloc][kl], sact[16 + nloc][kl + 1] };
        v2f a2 = { sact[32 + nloc][kl], sact[32 + nloc][kl + 1] };
        v2f a3 = { sact[48 + nloc][kl], sact[48 + nloc][kl + 1] };
        a0c = __builtin_amdgcn_wmma_f32_16x16x4_f32(false, a0, false, bf, (short)0, a0c, false, false);
        a1c = __builtin_amdgcn_wmma_f32_16x16x4_f32(false, a1, false, bf, (short)0, a1c, false, false);
        a2c = __builtin_amdgcn_wmma_f32_16x16x4_f32(false, a2, false, bf, (short)0, a2c, false, false);
        a3c = __builtin_amdgcn_wmma_f32_16x16x4_f32(false, a3, false, bf, (short)0, a3c, false, false);
    }
    float* pp = partial + (long)blk * 8192;
#pragma unroll
    for (int v = 0; v < 8; ++v) {
        int mrow = v + 8 * kh;
        pp[( 0 + mrow) * 128 + n] = a0c[v];
        pp[(16 + mrow) * 128 + n] = a1c[v];
        pp[(32 + mrow) * 128 + n] = a2c[v];
        pp[(48 + mrow) * 128 + n] = a3c[v];
    }
}

__global__ void reduce1(const float* __restrict__ partial, const float* __restrict__ bias,
                        float* __restrict__ out) {
    int i = blockIdx.x * 256 + threadIdx.x;          // 8192 = 64*128
    float s = 0.f;
    for (int k = 0; k < 512; ++k) s += partial[(long)k * 8192 + i];
    out[i] = lrelu(s + bias[i & 127]);
}

// ---------------------------------------------------------------------------
// Small dense: (64 x 128) @ (128 x 128) + bias, LeakyReLU.  One block, fp32 WMMA.
// ---------------------------------------------------------------------------
__global__ __launch_bounds__(256) void denseS(const float* __restrict__ act,
                                              const float* __restrict__ w,
                                              const float* __restrict__ bias,
                                              float* __restrict__ out)
{
    __shared__ __align__(16) float sact[64][132];
    const int tid = threadIdx.x, wave = tid >> 5, lane = tid & 31;
    for (int i = tid; i < 64 * 32; i += 256) {
        int mrow = i >> 5, kk4 = (i & 31) * 4;
        *(f32x4*)&sact[mrow][kk4] = *(const f32x4*)(act + mrow * 128 + kk4);
    }
    __syncthreads();
    const int nloc = lane & 15, kh = lane >> 4;
    const int n = wave * 16 + nloc;
    v8f a0c = {}, a1c = {}, a2c = {}, a3c = {};
    for (int kk = 0; kk < 128; kk += 4) {
        int k = kk + 2 * kh;
        v2f bf = { w[k * 128 + n], w[(k + 1) * 128 + n] };
        v2f a0 = { sact[ 0 + nloc][k], sact[ 0 + nloc][k + 1] };
        v2f a1 = { sact[16 + nloc][k], sact[16 + nloc][k + 1] };
        v2f a2 = { sact[32 + nloc][k], sact[32 + nloc][k + 1] };
        v2f a3 = { sact[48 + nloc][k], sact[48 + nloc][k + 1] };
        a0c = __builtin_amdgcn_wmma_f32_16x16x4_f32(false, a0, false, bf, (short)0, a0c, false, false);
        a1c = __builtin_amdgcn_wmma_f32_16x16x4_f32(false, a1, false, bf, (short)0, a1c, false, false);
        a2c = __builtin_amdgcn_wmma_f32_16x16x4_f32(false, a2, false, bf, (short)0, a2c, false, false);
        a3c = __builtin_amdgcn_wmma_f32_16x16x4_f32(false, a3, false, bf, (short)0, a3c, false, false);
    }
    float bn_ = bias[n];
#pragma unroll
    for (int v = 0; v < 8; ++v) {
        int mrow = v + 8 * kh;
        out[( 0 + mrow) * 128 + n] = lrelu(a0c[v] + bn_);
        out[(16 + mrow) * 128 + n] = lrelu(a1c[v] + bn_);
        out[(32 + mrow) * 128 + n] = lrelu(a2c[v] + bn_);
        out[(48 + mrow) * 128 + n] = lrelu(a3c[v] + bn_);
    }
}

__global__ void outk(const float* __restrict__ act, const float* __restrict__ w,
                     const float* __restrict__ bias, float* __restrict__ out) {
    int t = threadIdx.x;               // 128 = 64 batches x 2 outputs
    int brow = t >> 1, j = t & 1;
    float s = bias[j];
    for (int k = 0; k < 128; ++k) s += act[brow * 128 + k] * w[k * 2 + j];
    out[t] = s;
}

// ---------------------------------------------------------------------------
extern "C" void kernel_launch(void* const* d_in, const int* in_sizes, int n_in,
                              void* d_out, int out_size, void* d_ws, size_t ws_size,
                              hipStream_t stream)
{
    (void)in_sizes; (void)n_in; (void)out_size; (void)ws_size;
    const float* x     = (const float*)d_in[0];
    const float* wx[3] = {(const float*)d_in[1],  (const float*)d_in[8],  (const float*)d_in[15]};
    const float* wh[3] = {(const float*)d_in[2],  (const float*)d_in[9],  (const float*)d_in[16]};
    const float* bb[3] = {(const float*)d_in[3],  (const float*)d_in[10], (const float*)d_in[17]};
    const float* gg[3] = {(const float*)d_in[4],  (const float*)d_in[11], (const float*)d_in[18]};
    const float* bt[3] = {(const float*)d_in[5],  (const float*)d_in[12], (const float*)d_in[19]};
    const float* mu[3] = {(const float*)d_in[6],  (const float*)d_in[13], (const float*)d_in[20]};
    const float* vv[3] = {(const float*)d_in[7],  (const float*)d_in[14], (const float*)d_in[21]};
    const float* w_d1 = (const float*)d_in[22];
    const float* b_d1 = (const float*)d_in[23];
    const float* w_d2 = (const float*)d_in[24];
    const float* b_d2 = (const float*)d_in[25];
    const float* w_d3 = (const float*)d_in[26];
    const float* b_d3 = (const float*)d_in[27];
    const float* w_o  = (const float*)d_in[28];
    const float* b_o  = (const float*)d_in[29];
    float* out = (float*)d_out;

    char* ws = (char*)d_ws;
    size_t off = 0;
    auto take = [&](size_t bytes) -> char* {
        char* p = ws + off;
        off = (off + bytes + 255) & ~(size_t)255;
        return p;
    };
    unsigned short* bfr[3];
    for (int l = 0; l < 3; ++l) bfr[l] = (unsigned short*)take(3 * 32 * 16 * 2);
    const size_t HC = (size_t)64 * 128 * 128 * 4 * 4;          // h/c state, f32
    float* hA = (float*)take(HC);
    float* hB = (float*)take(HC);
    float* cS = (float*)take(HC);
    unsigned short* seq1 = (unsigned short*)take((size_t)64 * 5 * 128 * 128 * 4 * 2); // bf16
    unsigned short* seq2 = (unsigned short*)take((size_t)64 * 5 * 128 * 128 * 4 * 2); // bf16
    float* fin3 = (float*)take(HC);
    float* act1 = (float*)take(64 * 128 * 4);
    float* act2 = (float*)take(64 * 128 * 4);
    float* act3 = (float*)take(64 * 128 * 4);
    float* part = (float*)take((size_t)512 * 64 * 128 * 4);

    const int cins[3] = {3, 4, 4};
    for (int l = 0; l < 3; ++l)
        pack_bfrag<<<6, 256, 0, stream>>>(wx[l], wh[l], cins[l], bfr[l]);

    dim3 cg(8, 16, 64);
    const long seqT = (long)128 * 128 * 4;     // per-t elements in seq buffers
    const long seqB = 5 * seqT;                // per-batch elements

    // ---- layer 1 (input f32, CIN=3) ----
    hipMemsetAsync(hA, 0, HC, stream);
    hipMemsetAsync(cS, 0, HC, stream);
    {
        float *hi = hA, *ho = hB;
        for (int t = 0; t < 5; ++t) {
            convlstm_cell<3, false><<<cg, 256, 0, stream>>>(
                x + (long)t * 128 * 128 * 3, (long)5 * 128 * 128 * 3,
                hi, ho, cS,
                seq1 + (long)t * seqT, seqB,
                bfr[0], bb[0], gg[0], bt[0], mu[0], vv[0], 1);
            float* tmp = hi; hi = ho; ho = tmp;
        }
    }
    // ---- layer 2 (input bf16 seq1, CIN=4) ----
    hipMemsetAsync(hA, 0, HC, stream);
    hipMemsetAsync(cS, 0, HC, stream);
    {
        float *hi = hA, *ho = hB;
        for (int t = 0; t < 5; ++t) {
            convlstm_cell<4, false><<<cg, 256, 0, stream>>>(
                seq1 + (long)t * seqT, seqB,
                hi, ho, cS,
                seq2 + (long)t * seqT, seqB,
                bfr[1], bb[1], gg[1], bt[1], mu[1], vv[1], 1);
            float* tmp = hi; hi = ho; ho = tmp;
        }
    }
    // ---- layer 3 (input bf16 seq2, CIN=4; only last-t BN output, f32) ----
    hipMemsetAsync(hA, 0, HC, stream);
    hipMemsetAsync(cS, 0, HC, stream);
    {
        float *hi = hA, *ho = hB;
        for (int t = 0; t < 5; ++t) {
            convlstm_cell<4, true><<<cg, 256, 0, stream>>>(
                seq2 + (long)t * seqT, seqB,
                hi, ho, cS,
                fin3, (long)128 * 128 * 4,
                bfr[2], bb[2], gg[2], bt[2], mu[2], vv[2], t == 4 ? 1 : 0);
            float* tmp = hi; hi = ho; ho = tmp;
        }
    }

    // ---- dense head ----
    dense1<<<512, 256, 0, stream>>>(fin3, w_d1, part);
    reduce1<<<32, 256, 0, stream>>>(part, b_d1, act1);
    denseS<<<1, 256, 0, stream>>>(act1, w_d2, b_d2, act2);
    denseS<<<1, 256, 0, stream>>>(act2, w_d3, b_d3, act3);
    outk<<<1, 128, 0, stream>>>(act3, w_o, b_o, out);
}
